// erode_generation_18425409700384
// MI455X (gfx1250) — 8-bit matrix operand order repaired and checked by reading, compiled, not run on hardware
//
#include <hip/hip_runtime.h>

typedef __attribute__((ext_vector_type(8))) int v8i;

#define IMGW   1024
#define THRESH 121
#define SST    144
#define SGRP   36
#define SROWS  80
#define HST    84

__global__ __launch_bounds__(256) void erode_wmma_kernel(const float* __restrict__ in,
                                                         int* __restrict__ out)
{
    const int blk  = blockIdx.x;
    const int img  = blk >> 7;
    const int rblk = (blk >> 3) & 15;
    const int cblk = blk & 7;
    const int R0 = rblk * 64;
    const int C0 = cblk * 128;

    __shared__ unsigned char stage[SROWS * SST + 64];
    __shared__ unsigned char hbuf[8][16 * HST + 64];

    const int tid = threadIdx.x;
    const float* __restrict__ imgp = in + (size_t)img * (IMGW * IMGW);

    const bool interior = (rblk >= 1) && (rblk <= 14) && (cblk >= 1) && (cblk <= 6);
    if (interior) {
        const float* __restrict__ bp = imgp + (R0 - 5) * IMGW + (C0 - 8);
        for (int g = tid; g < SROWS * SGRP; g += 256) {
            const int sr  = g / SGRP;
            const int sc4 = (g - sr * SGRP) * 4;
            unsigned w = 0;
            if (sr < 74) {
                const float4 f = *(const float4*)(bp + sr * IMGW + sc4);
                w = (unsigned)f.x | ((unsigned)f.y << 8) |
                    ((unsigned)f.z << 16) | ((unsigned)f.w << 24);
            }
            *(unsigned*)&stage[sr * SST + sc4] = w;
        }
    } else {
        for (int i = tid; i < SROWS * SST; i += 256) {
            const int sr = i / SST;
            const int sc = i - sr * SST;
            const int r  = R0 - 5 + sr;
            const int c  = C0 - 8 + sc;
            unsigned char v = 0;
            if (sr < 74 && (unsigned)r < IMGW && (unsigned)c < IMGW)
                v = (imgp[r * IMGW + c] != 0.0f) ? 1u : 0u;
            stage[i] = v;
        }
    }
    __syncthreads();

    const int wave = tid >> 5;
    const int lane = tid & 31;
    const int n    = lane & 15;
    const int hi   = lane >> 4;
    const int tx   = wave;

    v8i Bh, Av;
#pragma unroll
    for (int g = 0; g < 8; ++g) {
        unsigned wb = 0, wa = 0;
#pragma unroll
        for (int b = 0; b < 4; ++b) {
            const int kb = ((g >> 1) << 4) + ((g & 1) << 2) + b + (hi << 3);
            const int ka = ((g >= 4) ? 32 : 0) + ((g & 3) << 2) + b + (hi << 4);
            if (kb >= n + 3 && kb <= n + 13) wb |= (1u << (8 * b));
            if (ka >= n && ka <= n + 10)     wa |= (1u << (8 * b));
        }
        Bh[g] = (int)wb;
        Av[g] = (int)wa;
    }

    unsigned char* __restrict__ hb = hbuf[wave];

    for (int c = 0; c < 5; ++c) {
        const unsigned char* rp = &stage[(c * 16 + n) * SST + tx * 16];
        v8i A;
#pragma unroll
        for (int g = 0; g < 8; ++g) {
            const int k0 = ((g >> 1) << 4) + ((g & 1) << 2) + (hi << 3);
            A[g] = *(const int*)(rp + k0);
        }
        v8i Cz = {};
        v8i D = __builtin_amdgcn_wmma_i32_16x16x64_iu8(false, A, false, Bh, Cz, false, false);
#pragma unroll
        for (int g = 0; g < 8; ++g)
            hb[n * HST + c * 16 + (hi << 3) + g] = (unsigned char)D[g];
    }

    for (int ty = 0; ty < 4; ++ty) {
        v8i B2;
#pragma unroll
        for (int g = 0; g < 8; ++g) {
            const int k0 = ((g >= 4) ? 32 : 0) + ((g & 3) << 2) + (hi << 4);
            B2[g] = *(const int*)(hb + n * HST + ty * 16 + k0);
        }
        v8i Cz = {};
        v8i D2 = __builtin_amdgcn_wmma_i32_16x16x64_iu8(false, Av, false, B2, Cz, false, false);

        int* __restrict__ op = out + (size_t)img * (IMGW * IMGW)
                             + (size_t)(R0 + ty * 16) * IMGW + (C0 + tx * 16) + n;
#pragma unroll
        for (int g = 0; g < 8; ++g)
            op[(size_t)(g + (hi << 3)) * IMGW] = (D2[g] >= THRESH) ? 1 : 0;
    }
}

extern "C" void kernel_launch(void* const* d_in, const int* in_sizes, int n_in,
                              void* d_out, int out_size, void* d_ws, size_t ws_size,
                              hipStream_t stream) {
    (void)in_sizes; (void)n_in; (void)d_ws; (void)ws_size; (void)out_size;
    const float* in = (const float*)d_in[0];
    int* out = (int*)d_out;
    const int blocks = 64 * 16 * 8;
    erode_wmma_kernel<<<blocks, 256, 0, stream>>>(in, out);
}
